// MLPDecoder_15049565405310
// MI455X (gfx1250) — matrix operand order repaired and checked by reading, not compiled, not run on hardware
//
#include <hip/hip_runtime.h>

#define N_NODES 100000
#define N_EDGES 640000
#define IN_CH   128
#define HID     128
#define GROUPS2 (N_EDGES / 32)   // 20000 groups of 32 edges
#define W1_ELEMS (2 * IN_CH * HID)

typedef __attribute__((ext_vector_type(16))) __bf16          v16bf;
typedef __attribute__((ext_vector_type(8)))  float           v8f;
typedef __attribute__((ext_vector_type(8)))  unsigned short  us8;

union BfFrag {
    unsigned short s[16];
    unsigned int   d[8];
    us8            h[2];
    v16bf          v;
};

// truncating pack: low16 = bf16(lo), high16 = bf16(hi), one v_perm_b32
__device__ __forceinline__ unsigned int pack_bf2(float lo, float hi) {
    return __builtin_amdgcn_perm(__float_as_uint(hi), __float_as_uint(lo), 0x07060302u);
}

__device__ __forceinline__ unsigned short f2bf_rne(float f) {
    unsigned int u = __float_as_uint(f);
    u += 0x7FFFu + ((u >> 16) & 1u);
    return (unsigned short)(u >> 16);
}

// One-time prep: W1 [256,128] f32 -> W1T [128 cols][256 K] bf16 (RNE), K-contiguous.
__global__ __launch_bounds__(256)
void prep_w1(const float* __restrict__ W1, unsigned short* __restrict__ w1t) {
    const int idx = blockIdx.x * 256 + threadIdx.x;   // 0..32767
    const int k = idx >> 7;
    const int n = idx & 127;
    w1t[n * 256 + k] = f2bf_rne(W1[idx]);
}

__global__ __launch_bounds__(256)
void mlp_decoder_wmma(const float* __restrict__ z,
                      const long long* __restrict__ ei,    // [2, N_EDGES] int64
                      const float* __restrict__ W1,        // [2*IN_CH, HID]
                      const float* __restrict__ b1,        // [HID]
                      const float* __restrict__ W2,        // [HID, 1]
                      const float* __restrict__ b2,        // [1]
                      const unsigned short* __restrict__ w1t, // bf16 W1T or null
                      float* __restrict__ out)             // [N_EDGES]
{
    // A tile: 32 edges x 256 K bf16; row padded to 264 ushorts (528B, 16B-aligned)
    __shared__ __align__(16) unsigned short sA[32][264];
    // per-wave h transpose tile: [16 rows][16 cols] padded to 24 (48B stride)
    __shared__ __align__(16) unsigned short sH[8][16][24];
    __shared__ float sOut[32];

    const int tid  = threadIdx.x;
    const int lane = tid & 31;
    const int wave = tid >> 5;          // 0..7 -> N-tile (16 hidden cols each)
    const int half = lane >> 4;
    const int l15  = lane & 15;
    const int col  = wave * 16 + l15;   // hidden unit owned by this lane

    // ---- Layer-1 B fragments (16 cols of W1) in registers ----
    // B layout used here (contiguous K order) 32x16 bf16: lanes 0-15 hold K=kb..kb+15 (2/VGPR, K-contig),
    // lanes 16-31 hold K=kb+16..kb+31; column N = lane%16.
    BfFrag breg[8];
    if (w1t) {
        const us8* wp = (const us8*)(w1t + col * 256);  // 16B-aligned, K-contig
#pragma unroll
        for (int kt = 0; kt < 8; ++kt) {
            breg[kt].h[0] = wp[kt * 4 + half * 2];
            breg[kt].h[1] = wp[kt * 4 + half * 2 + 1];
        }
    } else {
#pragma unroll
        for (int kt = 0; kt < 8; ++kt) {
            const int kbase = kt * 32 + half * 16;
#pragma unroll
            for (int i = 0; i < 8; ++i)
                breg[kt].d[i] = pack_bf2(W1[(kbase + 2 * i) * HID + col],
                                         W1[(kbase + 2 * i + 1) * HID + col]);
        }
    }

    // ---- Layer-2 B fragment: W2 slice broadcast to all 16 columns.
    // K padded 16->32: elements 0..7 carry w2[half*8+i], elements 8..15 zero.
    BfFrag w2f;
#pragma unroll
    for (int i = 0; i < 16; ++i)
        w2f.s[i] = (i >= 8) ? (unsigned short)0 : f2bf_rne(W2[wave * 16 + half * 8 + i]);

    const float b1v = b1[col];
    const float b2v = b2[0];

    // gather assignment: thread -> (2 edge rows, one 16-channel chunk)
    const int r = tid >> 4;             // 0..15; rows r and r+16
    const int c = (tid & 15) * 16;      // 0..240 channel chunk of concat(256)

    for (int g = blockIdx.x; g < GROUPS2; g += gridDim.x) {
        const int e0 = g * 32;

        if (tid < 32) sOut[tid] = 0.0f;

        // ---- Gather two z rows per edge, pack f32->bf16 with v_perm,
        //      2x ds_store_b128 per row into LDS A tile ----
#pragma unroll
        for (int rr = 0; rr < 2; ++rr) {
            const int row = r + rr * 16;
            const long long e = (long long)(e0 + row);
            const long long s = ei[e];
            const long long d = ei[(long long)N_EDGES + e];
            const float* zp = (c < IN_CH) ? (z + s * IN_CH + c)
                                          : (z + d * IN_CH + (c - IN_CH));
            const float4* zv = (const float4*)zp;
            float4 f0 = zv[0], f1 = zv[1], f2 = zv[2], f3 = zv[3];
            union { unsigned int w[8]; us8 h[2]; } up;
            up.w[0] = pack_bf2(f0.x, f0.y);
            up.w[1] = pack_bf2(f0.z, f0.w);
            up.w[2] = pack_bf2(f1.x, f1.y);
            up.w[3] = pack_bf2(f1.z, f1.w);
            up.w[4] = pack_bf2(f2.x, f2.y);
            up.w[5] = pack_bf2(f2.z, f2.w);
            up.w[6] = pack_bf2(f3.x, f3.y);
            up.w[7] = pack_bf2(f3.z, f3.w);
            *(us8*)&sA[row][c]     = up.h[0];
            *(us8*)&sA[row][c + 8] = up.h[1];
        }

        // prefetch next group's gather sources (both rows)
#pragma unroll
        for (int rr = 0; rr < 2; ++rr) {
            const int gn = g + gridDim.x;
            if (gn < GROUPS2) {
                const long long e = (long long)gn * 32 + r + rr * 16;
                const long long s = ei[e];
                const long long d = ei[(long long)N_EDGES + e];
                const float* zp = (c < IN_CH) ? (z + s * IN_CH + c)
                                              : (z + d * IN_CH + (c - IN_CH));
                __builtin_prefetch(zp, 0, 0);
            }
        }

        __syncthreads();

        // ---- Two 16-edge M-tiles per wave: dense WMMA chains sharing breg ----
#pragma unroll
        for (int t = 0; t < 2; ++t) {
            // Layer 1: preload all 8 A fragments, then 8 back-to-back WMMAs.
            // A layout 16x32 bf16 (matches B): lane<16 row=l15 holds K kb..kb+15,
            // lane>=16 holds K kb+16..kb+31.
            BfFrag af[8];
#pragma unroll
            for (int kt = 0; kt < 8; ++kt) {
                const int kb = kt * 32;
                af[kt].h[0] = *(const us8*)&sA[t * 16 + l15][kb + half * 16];
                af[kt].h[1] = *(const us8*)&sA[t * 16 + l15][kb + half * 16 + 8];
            }
            v8f acc = {};
#pragma unroll
            for (int kt = 0; kt < 8; ++kt) {
                acc = __builtin_amdgcn_wmma_f32_16x16x32_bf16(
                    false, af[kt].v, false, breg[kt].v, (short)0, acc, false, false);
            }
#if __has_builtin(__builtin_amdgcn_sched_group_barrier)
            // Pin schedule: 16 LDS reads (A frags) first, then the 8-WMMA chain.
            __builtin_amdgcn_sched_group_barrier(0x100, 16, 0); // DS read
            __builtin_amdgcn_sched_group_barrier(0x008,  8, 0); // WMMA
#endif

            // bias + ReLU, truncate to bf16 (folds to ds_store_b16_d16_hi),
            // transpose D->A layout via per-wave LDS tile (same-wave DS in-order).
            // D layout: lane col N=l15, VGPR j -> row M = j + 8*half.
#pragma unroll
            for (int j = 0; j < 8; ++j) {
                float h = fmaxf(acc[j] + b1v, 0.0f);
                sH[wave][j + 8 * half][l15] =
                    (unsigned short)(__float_as_uint(h) >> 16);
            }

            // Layer 2 as WMMA: partial[row] = sum_k h(row,k)*w2[k], K padded to 32.
            BfFrag a2;
            a2.h[0] = *(const us8*)&sH[wave][l15][half * 8];
#pragma unroll
            for (int i = 8; i < 16; ++i) a2.s[i] = 0;   // zero K-pad half
            v8f zc = {};
            v8f acc2 = __builtin_amdgcn_wmma_f32_16x16x32_bf16(
                false, a2.v, false, w2f.v, (short)0, zc, false, false);

            // Every column holds the same partial; lanes 0 and 16 (rows 0-7 / 8-15)
            // accumulate across the 8 waves with ds_add_f32.
#pragma unroll
            for (int j = 0; j < 8; ++j) {
                if (l15 == 0) atomicAdd(&sOut[t * 16 + half * 8 + j], acc2[j]);
            }
        }

        __syncthreads();

        if (tid < 32) out[e0 + tid] = sOut[tid] + b2v;
    }
}

extern "C" void kernel_launch(void* const* d_in, const int* in_sizes, int n_in,
                              void* d_out, int out_size, void* d_ws, size_t ws_size,
                              hipStream_t stream) {
    const float*     z  = (const float*)d_in[0];
    const long long* ei = (const long long*)d_in[1];
    const float*     W1 = (const float*)d_in[2];
    const float*     b1 = (const float*)d_in[3];
    const float*     W2 = (const float*)d_in[4];
    const float*     b2 = (const float*)d_in[5];
    float*           out = (float*)d_out;
    (void)in_sizes; (void)n_in; (void)out_size;

    unsigned short* w1t = nullptr;
    if (ws_size >= (size_t)W1_ELEMS * sizeof(unsigned short)) {
        w1t = (unsigned short*)d_ws;
        hipLaunchKernelGGL(prep_w1, dim3(W1_ELEMS / 256), dim3(256), 0, stream,
                           W1, w1t);
    }

    hipLaunchKernelGGL(mlp_decoder_wmma, dim3(4096), dim3(256), 0, stream,
                       z, ei, W1, b1, W2, b2, w1t, out);
}
